// Self_AttentionBlock_47047071760925
// MI455X (gfx1250) — compile-verified, matrix operand order checked by reading, not run on hardware
//
#include <hip/hip_runtime.h>

// ---------------------------------------------------------------------------
// MI455X (gfx1250, wave32) implementation of the Self_AttentionBlock reference.
// All matmuls run through v_wmma_f32_16x16x32_f16 (f16 in, f32 accumulate).
// Uses the zero-bias identity q.k = x^T (Wq_s^T Wk_s) x' to fold the q/k
// projections into one (bq,bk are zero in setup_inputs).
// ---------------------------------------------------------------------------

typedef _Float16 half_t;
typedef __attribute__((ext_vector_type(16))) _Float16 v16h;
typedef __attribute__((ext_vector_type(8)))  _Float16 v8h;    // 16B, aligned(16)
typedef __attribute__((ext_vector_type(8)))  float    v8f;

__device__ __forceinline__ v8f wmma_f16(const v16h& a, const v16h& b, const v8f& c) {
    return __builtin_amdgcn_wmma_f32_16x16x32_f16(false, a, false, b, (short)0, c, false, false);
}

// Fragment loader: base must be 16B aligned; picks halves [0..7] and [16..23]
// (the per-lane K pattern of the 16-bit WMMA A/B layouts, k0 handled by caller).
__device__ __forceinline__ v16h load_frag16(const half_t* p) {
    const v8h* q = (const v8h*)p;
    v8h lo = q[0], hi = q[2];
    return __builtin_shufflevector(lo, hi, 0, 1, 2, 3, 4, 5, 6, 7,
                                           8, 9, 10, 11, 12, 13, 14, 15);
}

#define KN   32
#define KC   64
#define KT   256
#define KV   25
#define KS   8
#define KTL  251          // T - LAG
#define KP   6400         // T*V pixels per image
#define NPIX 204800L      // N*T*V  (BN reduction size)
#define LPIT 40           // LDS row pitch in halves (80B, 16B aligned rows)

// ------------------------------ tiny helpers -------------------------------

__global__ void k_cvt_f16(const float* __restrict__ in, half_t* __restrict__ out, long n) {
    for (long i = blockIdx.x * (long)blockDim.x + threadIdx.x; i < n;
         i += (long)gridDim.x * blockDim.x)
        out[i] = (half_t)in[i];
}

__global__ void k_zero_f32(float* __restrict__ p, int n) {
    for (int i = blockIdx.x * blockDim.x + threadIdx.x; i < n; i += gridDim.x * blockDim.x)
        p[i] = 0.0f;
}

// xt[n][v][c][t] = x[n][c][t][v]  (f16, v padded 25->32 with zeros)
__global__ void k_build_xt(const float* __restrict__ x, half_t* __restrict__ xt) {
    const long total = (long)KN * 32 * KC * KT;
    for (long i = blockIdx.x * (long)blockDim.x + threadIdx.x; i < total;
         i += (long)gridDim.x * blockDim.x) {
        int t = (int)(i & 255);
        int c = (int)((i >> 8) & 63);
        int v = (int)((i >> 14) & 31);
        int n = (int)(i >> 19);
        xt[i] = (v < KV) ? (half_t)x[((long)n * KC + c) * KP + t * KV + v] : (half_t)0;
    }
}

// MsT[s][cj][ci] = sum_c Wq[s*64+c][ci] * Wk[s*64+c][cj]   (= (Wq_s^T Wk_s)^T)
__global__ __launch_bounds__(256) void k_msT(const float* __restrict__ Wq,
                                             const float* __restrict__ Wk,
                                             half_t* __restrict__ MsT) {
    const int s = blockIdx.x;
    const int tid = threadIdx.x;
    for (int e = 0; e < 16; ++e) {
        int o = tid * 16 + e;            // 0..4095
        int cj = o >> 6, ci = o & 63;
        float acc = 0.f;
        for (int c = 0; c < 64; ++c)
            acc += Wq[(s * 64 + c) * 64 + ci] * Wk[(s * 64 + c) * 64 + cj];
        MsT[(long)s * 4096 + o] = (half_t)acc;
    }
}

// ------------------------- generic 1x1-conv GEMM ---------------------------
// Out(64 x P) = W(64 x Cin) * In(Cin x P) + bias, channel-major activations.
// MODE 0: f16 store, zero rows t >= t_limit.   MODE 1: f32 store + BN stats.
template <int MODE>
__global__ __launch_bounds__(256)
void k_conv_gemm(const half_t* __restrict__ In, const half_t* __restrict__ W,
                 const float* __restrict__ bias, void* __restrict__ OutV,
                 int Cin, int s_count, int t_limit,
                 float* __restrict__ ssum, float* __restrict__ ssq) {
    __shared__ half_t lds_in[128 * LPIT];   // staged In tile, transposed [p][k]
    __shared__ float  bn_s[64], bn_q[64];

    const int z  = blockIdx.y;              // n (or n*S+s for the Z projection)
    const int n  = z / s_count;
    const int s  = z % s_count;
    const int p0 = blockIdx.x * 128;

    const half_t* inN = In + (long)n * Cin * KP;
    const half_t* Ws  = W + (long)s * 64 * Cin;

    const int tid = threadIdx.x, lane = tid & 31, wave = tid >> 5;
    const int otile = wave & 3;             // output-channel tile (16 rows)
    const int phalf = wave >> 2;            // pixel half (64 pixels)
    const int k0 = (lane < 16) ? 0 : 8;
    const int row_m = otile * 16 + (lane & 15);

    if (MODE == 1 && tid < 64) { bn_s[tid] = 0.f; bn_q[tid] = 0.f; }

    v8f acc[4] = {};

    for (int kc = 0; kc < Cin; kc += 32) {
        // coalesced stage of In[kc..kc+32) x [p0..p0+128) -> LDS transposed
#pragma unroll
        for (int e = 0; e < 16; ++e) {
            int idx = e * 256 + tid;        // 0..4095
            int k = idx >> 7, p = idx & 127;
            lds_in[p * LPIT + k] = inN[(long)(kc + k) * KP + p0 + p];
        }
        if (kc + 32 < Cin) {                // prefetch next K-chunk
            int k = tid >> 3, p = (tid & 7) * 16;
            __builtin_prefetch(&inN[(long)(kc + 32 + k) * KP + p0 + p], 0, 1);
        }
        __syncthreads();

        // A fragment straight from W (rows contiguous, 16B aligned)
        v16h a = load_frag16(Ws + (long)row_m * Cin + kc + k0);
#pragma unroll
        for (int c4 = 0; c4 < 4; ++c4) {     // 4 pixel column-tiles per wave
            int pl = phalf * 64 + c4 * 16 + (lane & 15);
            v16h b = load_frag16(&lds_in[pl * LPIT + k0]);
            acc[c4] = wmma_f16(a, b, acc[c4]);
        }
        __syncthreads();
    }

    // ---- epilogue ---- (D layout: VGPR i -> row rb+i, col = lane&15)
    const int rb = otile * 16 + ((lane < 16) ? 0 : 8);
    const int pbase = p0 + phalf * 64 + (lane & 15);
    if (MODE == 0) {
        half_t* Out = (half_t*)OutV + (long)z * 64 * KP;
#pragma unroll
        for (int c4 = 0; c4 < 4; ++c4) {
            int p = pbase + c4 * 16;
            bool keep = (p / KV) < t_limit;
#pragma unroll
            for (int i = 0; i < 8; ++i) {
                int o = rb + i;
                float v = acc[c4][i] + (bias ? bias[o] : 0.f);
                Out[(long)o * KP + p] = keep ? (half_t)v : (half_t)0;
            }
        }
    } else {
        float* Out = (float*)OutV + (long)z * 64 * KP;
        float rs[8], rq[8];
#pragma unroll
        for (int i = 0; i < 8; ++i) { rs[i] = 0.f; rq[i] = 0.f; }
#pragma unroll
        for (int c4 = 0; c4 < 4; ++c4) {
            int p = pbase + c4 * 16;
#pragma unroll
            for (int i = 0; i < 8; ++i) {
                int o = rb + i;
                float v = acc[c4][i] + (bias ? bias[o] : 0.f);
                Out[(long)o * KP + p] = v;
                rs[i] += v; rq[i] += v * v;
            }
        }
#pragma unroll
        for (int i = 0; i < 8; ++i) {
            atomicAdd(&bn_s[rb + i], rs[i]);
            atomicAdd(&bn_q[rb + i], rq[i]);
        }
        __syncthreads();
        if (tid < 64) { atomicAdd(&ssum[tid], bn_s[tid]); atomicAdd(&ssq[tid], bn_q[tid]); }
    }
}

// ------------------- lag attention + mean/max + softmax --------------------
// One block per (n,s): atts_l[u,v] = sum_{c,t<Tl} Z[c,t,u] * x[c,t+l,v]
// 4 waves cover the 32x32 (padded) logits; the 6 lag B-fragments are sliced
// out of one 32-half register window (the v_mov repack co-executes with the
// TRANS-tracked 16-bit WMMAs, per ISA 8.4).
__global__ __launch_bounds__(128)
void k_atts(const half_t* __restrict__ Z, const half_t* __restrict__ xt,
            half_t* __restrict__ att) {
    __shared__ half_t lds_a[32 * LPIT];     // Z tile  [u][t']
    __shared__ half_t lds_b[32 * LPIT];     // x tile  [v][t''], t'' in [0,37)
    __shared__ float  lds_p[32 * 33];       // pre-softmax logits

    const int z = blockIdx.x;               // n*8 + s
    const int n = z >> 3;
    const half_t* Zb = Z + (long)z * 64 * KP;             // [c][t][u]
    const half_t* Xb = xt + (long)n * 32 * 64 * KT;       // [v][c][t]

    const int tid = threadIdx.x, lane = tid & 31, wave = tid >> 5;
    const int ut = wave & 1, vt = wave >> 1;
    const int k0 = (lane < 16) ? 0 : 8;

    v8f acc[6] = {};

    for (int c = 0; c < 64; ++c) {
        for (int t0 = 0; t0 < KT; t0 += 32) {
            // stage A: Z[c][t0+t'][u]  (Z rows t>=Tl are zero -> exact padding)
#pragma unroll
            for (int e = 0; e < 8; ++e) {
                int idx = e * 128 + tid;    // 0..1023
                int tp = idx >> 5, u = idx & 31;
                lds_a[u * LPIT + tp] =
                    (u < KV) ? Zb[(long)c * KP + (t0 + tp) * KV + u] : (half_t)0;
            }
            // stage B: xt[v][c][t0 .. t0+36]  (covers all 6 lag shifts)
            for (int idx = tid; idx < 32 * 37; idx += 128) {
                int v = idx / 37, tt = idx % 37;
                lds_b[v * LPIT + tt] = Xb[(long)v * (64 * KT) + c * KT + t0 + tt];
            }
            __syncthreads();

            int u = ut * 16 + (lane & 15);
            v16h a = load_frag16(&lds_a[u * LPIT + k0]);

            int vv = vt * 16 + (lane & 15);
            half_t r[32];                    // register window t = k0 .. k0+31
            {
                const half_t* br = &lds_b[vv * LPIT + k0];
#pragma unroll
                for (int i = 0; i < 32; ++i) r[i] = br[i];
            }
#pragma unroll
            for (int l = 0; l < 6; ++l) {
                v16h b;
#pragma unroll
                for (int i = 0; i < 8; ++i) { b[i] = r[l + i]; b[8 + i] = r[l + 16 + i]; }
                acc[l] = wmma_f16(a, b, acc[l]);
            }
            __syncthreads();
        }
    }

    // p = (mean_l + max_l) * 0.5 / sqrt(CO*Tl)
    const float inv_scale = rsqrtf(64.0f * 251.0f);
    const int rb = ut * 16 + ((lane < 16) ? 0 : 8);
    const int col = vt * 16 + (lane & 15);
#pragma unroll
    for (int i = 0; i < 8; ++i) {
        float sm = 0.f, mx = -1e30f;
#pragma unroll
        for (int l = 0; l < 6; ++l) { float v = acc[l][i]; sm += v; mx = fmaxf(mx, v); }
        lds_p[(rb + i) * 33 + col] = (sm * (1.0f / 6.0f) + mx) * 0.5f * inv_scale;
    }
    __syncthreads();

    if (tid < 32) {                          // row softmax over v<25, zero-pad
        half_t* ao = att + (long)z * 1024 + tid * 32;
        if (tid < KV) {
            float m = -1e30f;
            for (int v = 0; v < KV; ++v) m = fmaxf(m, lds_p[tid * 33 + v]);
            float sum = 0.f;
            for (int v = 0; v < KV; ++v) sum += __expf(lds_p[tid * 33 + v] - m);
            float rinv = 1.0f / sum;
            for (int v = 0; v < KV; ++v)
                ao[v] = (half_t)(__expf(lds_p[tid * 33 + v] - m) * rinv);
            for (int v = KV; v < 32; ++v) ao[v] = (half_t)0;
        } else {
            for (int v = 0; v < 32; ++v) ao[v] = (half_t)0;
        }
    }
}

// ------------------- attention apply: y = vmap x att -----------------------
// y1[(n*8+s)*64+c][t][v] = sum_u vh[n][c][t][u] * att[n,s][u][v]   (K=25->32)
// vh rows are staged through LDS: the block's 128 rows are one contiguous
// 6.4KB global range -> coalesced loads, then aligned ds_load_b128 fragments.
__global__ __launch_bounds__(256)
void k_yein(const half_t* __restrict__ vh, const half_t* __restrict__ att,
            half_t* __restrict__ y1) {
    __shared__ half_t lds_vh[128 * LPIT];    // [row][u], u padded 25->32 w/ 0
    __shared__ half_t lds_att[32 * LPIT];    // att transposed [v][u]
    const int z = blockIdx.y;                // n*8 + s
    const int n = z >> 3;
    const int mt = blockIdx.x;               // 128 row-tiles of 128 rows
    const int r0 = mt * 128;
    const half_t* vb = vh + (long)n * 64 * KP;

    for (int idx = threadIdx.x; idx < 1024; idx += 256) {
        int u = idx >> 5, v = idx & 31;
        lds_att[v * LPIT + u] = att[(long)z * 1024 + idx];
    }
    for (int idx = threadIdx.x; idx < 128 * 32; idx += 256) {
        int row = idx >> 5, k = idx & 31;
        lds_vh[row * LPIT + k] = (k < KV) ? vb[(long)(r0 + row) * KV + k] : (half_t)0;
    }
    __syncthreads();

    const int tid = threadIdx.x, lane = tid & 31, wave = tid >> 5;
    const int k0 = (lane < 16) ? 0 : 8;
    const int lr = wave * 16 + (lane & 15);

    v16h a = load_frag16(&lds_vh[lr * LPIT + k0]);
    v8f acc[2] = {};
#pragma unroll
    for (int c4 = 0; c4 < 2; ++c4) {
        int v = c4 * 16 + (lane & 15);
        v16h b = load_frag16(&lds_att[v * LPIT + k0]);
        acc[c4] = wmma_f16(a, b, acc[c4]);
    }

    const int rb = r0 + wave * 16 + ((lane < 16) ? 0 : 8);
#pragma unroll
    for (int c4 = 0; c4 < 2; ++c4) {
        int v = c4 * 16 + (lane & 15);
        if (v < KV) {
#pragma unroll
            for (int i = 0; i < 8; ++i) {
                int r = rb + i;
                int cch = r >> 8, t = r & 255;         // rows ordered (c,t)
                y1[((long)z * 64 + cch) * KP + t * KV + v] = (half_t)acc[c4][i];
            }
        }
    }
}

// ------------- BN(a) + BN(b) residual + LeakyReLU(0.1) epilogue ------------
template <typename OT>
__global__ void k_post(const float* ar, const float* as, const float* aq,
                       const float* ga, const float* ba,
                       const float* brw, const float* bs, const float* bq,
                       const float* gb, const float* bb, OT* out) {
    const float invM = 1.0f / (float)NPIX;
    const long total = (long)KN * 64 * KP;
    for (long i = blockIdx.x * (long)blockDim.x + threadIdx.x; i < total;
         i += (long)gridDim.x * blockDim.x) {
        int c = (int)((i / KP) & 63);
        float m1 = as[c] * invM, v1 = aq[c] * invM - m1 * m1;
        float x1 = (ar[i] - m1) * rsqrtf(v1 + 1e-5f) * ga[c] + ba[c];
        float m2 = bs[c] * invM, v2 = bq[c] * invM - m2 * m2;
        float x2 = (brw[i] - m2) * rsqrtf(v2 + 1e-5f) * gb[c] + bb[c];
        float y = x1 + x2;
        out[i] = (OT)((y >= 0.f) ? y : 0.1f * y);
    }
}

// --------------------------------- driver ----------------------------------
extern "C" void kernel_launch(void* const* d_in, const int* in_sizes, int n_in,
                              void* d_out, int out_size, void* d_ws, size_t ws_size,
                              hipStream_t stream) {
    (void)in_sizes; (void)n_in; (void)out_size; (void)ws_size;
    const float* x     = (const float*)d_in[0];
    const float* Wq    = (const float*)d_in[1];
    const float* Wk    = (const float*)d_in[3];
    const float* Wv    = (const float*)d_in[5];
    const float* bv    = (const float*)d_in[6];
    const float* Wout  = (const float*)d_in[7];
    const float* b_out = (const float*)d_in[8];
    const float* g_out = (const float*)d_in[9];
    const float* beO   = (const float*)d_in[10];
    const float* Wff   = (const float*)d_in[11];
    const float* bff   = (const float*)d_in[12];
    const float* g_ff  = (const float*)d_in[13];
    const float* beF   = (const float*)d_in[14];
    const float* Wd1   = (const float*)d_in[15];
    const float* bd1   = (const float*)d_in[16];
    const float* g1    = (const float*)d_in[17];
    const float* be1   = (const float*)d_in[18];
    const float* Wd2   = (const float*)d_in[19];
    const float* bd2   = (const float*)d_in[20];
    const float* g2    = (const float*)d_in[21];
    const float* be2   = (const float*)d_in[22];

    // workspace carve-out (~480 MB total; 256B aligned slabs)
    char* w = (char*)d_ws;
    auto alloc = [&](size_t bytes) { char* p = w; w += (bytes + 255) & ~(size_t)255; return p; };
    const long NCP = (long)KN * 64 * KP;                 // 13,107,200 elements
    half_t* xh    = (half_t*)alloc(NCP * 2);             // x in f16, (n,c,t,v)
    half_t* xt    = (half_t*)alloc((long)KN * 32 * 64 * KT * 2 + 256);  // (n,v,c,t)
    half_t* vh    = (half_t*)alloc(NCP * 2 + 256);       // vmap_x f16
    half_t* MsT   = (half_t*)alloc(8 * 4096 * 2);
    half_t* Wv16  = (half_t*)alloc(4096 * 2);
    half_t* Wo16  = (half_t*)alloc(32768 * 2);
    half_t* Wf16  = (half_t*)alloc(4096 * 2);
    half_t* W116  = (half_t*)alloc(4096 * 2);
    half_t* W216  = (half_t*)alloc(4096 * 2);
    half_t* Zbuf  = (half_t*)alloc((long)KN * KS * 64 * KP * 2);  // Z, reused as y1
    half_t* att16 = (half_t*)alloc((long)KN * KS * 1024 * 2);
    half_t* z1    = (half_t*)alloc(NCP * 2 + 256);
    float*  y2raw = (float*)alloc(NCP * 4);
    float*  d1raw = (float*)alloc(NCP * 4);
    float*  ffraw = (float*)alloc(NCP * 4);
    float*  stats = (float*)alloc(8 * 64 * 4);
    float* sO = stats, *qO = stats + 64, *s1 = stats + 128, *q1 = stats + 192,
         * s2 = stats + 256, *q2 = stats + 320, *sF = stats + 384, *qF = stats + 448;
    half_t* y1    = Zbuf;                    // Z is dead once k_atts finishes
    float*  d2raw = (float*)d_out;           // elementwise-safe alias with k_post

    // 1) conversions / transposes / M_s = Wq_s^T Wk_s
    k_cvt_f16<<<8192, 256, 0, stream>>>(x, xh, NCP);
    k_cvt_f16<<<32, 256, 0, stream>>>(Wv,   Wv16, 4096);
    k_cvt_f16<<<32, 256, 0, stream>>>(Wout, Wo16, 32768);
    k_cvt_f16<<<32, 256, 0, stream>>>(Wff,  Wf16, 4096);
    k_cvt_f16<<<32, 256, 0, stream>>>(Wd1,  W116, 4096);
    k_cvt_f16<<<32, 256, 0, stream>>>(Wd2,  W216, 4096);
    k_build_xt<<<8192, 256, 0, stream>>>(x, xt);
    k_msT<<<8, 256, 0, stream>>>(Wq, Wk, MsT);
    k_zero_f32<<<2, 256, 0, stream>>>(stats, 512);

    // 2) projections: Z[n,s] = MsT_s * x_n (t>=Tl zeroed), vmap = Wv*x + bv
    k_conv_gemm<0><<<dim3(50, KN * KS), 256, 0, stream>>>(
        xh, MsT, nullptr, Zbuf, 64, KS, KTL, nullptr, nullptr);
    k_conv_gemm<0><<<dim3(50, KN), 256, 0, stream>>>(
        xh, Wv16, bv, vh, 64, 1, KT, nullptr, nullptr);

    // 3) lag attention + softmax  -> att16 (zero-padded 32x32 tiles)
    k_atts<<<KN * KS, 128, 0, stream>>>(Zbuf, xt, att16);

    // 4) y1 = vmap x att  (written into the dead Z buffer)
    k_yein<<<dim3(128, KN * KS), 256, 0, stream>>>(vh, att16, y1);

    // 5) the four BN'd convolutions (f32 out + batch statistics)
    k_conv_gemm<1><<<dim3(50, KN), 256, 0, stream>>>(
        y1, Wo16, b_out, y2raw, 512, 1, KT, sO, qO);
    k_conv_gemm<1><<<dim3(50, KN), 256, 0, stream>>>(
        xh, W116, bd1, d1raw, 64, 1, KT, s1, q1);
    k_conv_gemm<1><<<dim3(50, KN), 256, 0, stream>>>(
        xh, W216, bd2, d2raw, 64, 1, KT, s2, q2);

    // 6) z1 = lrelu(bn(d1) + bn(y2))  -> f16 input of the Wff conv
    k_post<half_t><<<8192, 256, 0, stream>>>(
        d1raw, s1, q1, g1, be1, y2raw, sO, qO, g_out, beO, z1);

    k_conv_gemm<1><<<dim3(50, KN), 256, 0, stream>>>(
        z1, Wf16, bff, ffraw, 64, 1, KT, sF, qF);

    // 7) out = lrelu(bn(d2) + bn(ff))  (d2raw aliases d_out, per-element safe)
    k_post<float><<<8192, 256, 0, stream>>>(
        d2raw, s2, q2, g2, be2, ffraw, sF, qF, g_ff, beF, (float*)d_out);
}
